// ClassicalSelfAttention_65481071402132
// MI455X (gfx1250) — matrix operand order repaired and checked by reading, not compiled, not run on hardware
//
#include <hip/hip_runtime.h>
#include <hip/hip_bf16.h>

// ---------------------------------------------------------------------------
// ClassicalSelfAttention for MI455X (gfx1250, wave32, WMMA).
// All GEMMs run on v_wmma_f32_16x16x32_bf16; softmax + accumulation in fp32.
// ---------------------------------------------------------------------------

typedef __bf16 bf16_t;
typedef __attribute__((ext_vector_type(16))) __bf16 v16bf;
typedef __attribute__((ext_vector_type(8)))  __bf16 v8bf;
typedef __attribute__((ext_vector_type(8)))  float  v8f;

#define SEQ   4096
#define EMB   1024
#define BATCH 4
#define BQ    16      // queries per workgroup (attention)
#define BK    64      // key block per iteration
#define NWAVE 8       // waves per workgroup
#define DSL   128     // d-slice per wave (8 * 128 = 1024)
#define SLD   68      // padded LDS row stride (floats) to dodge bank conflicts

__device__ static inline v16bf frag_combine(v8bf lo, v8bf hi) {
  v16bf a;
#pragma unroll
  for (int i = 0; i < 8; ++i) { a[i] = lo[i]; a[i + 8] = hi[i]; }
  return a;
}

// A-fragment (16x32 bf16, MxK): lane m = lane%16, half h = lane/16.
// VGPR j<4 holds K = 2j+{0,1}+8h ; VGPR j>=4 holds K = 16+2(j-4)+{0,1}+8h.
// -> per lane: chunks [8h, 8h+8) and [16+8h, 16+8h+8) of row m.
__device__ static inline v16bf load_fragA(const bf16_t* base, int ld) {
  const int lane = threadIdx.x & 31;
  const int m = lane & 15, h = lane >> 4;
  const bf16_t* p = base + (size_t)m * ld + 8 * h;
  v8bf lo = *(const v8bf*)p;
  v8bf hi = *(const v8bf*)(p + 16);
  return frag_combine(lo, hi);
}

// B-fragment (32x16 bf16, KxN), loaded from row-major B^T (N rows, K cols):
// lane n = lane%16, half h = lane/16 -> K chunks [8h, 8h+8) and [16+8h, 16+8h+8).
__device__ static inline v16bf load_fragB(const bf16_t* baseT, int ld) {
  const int lane = threadIdx.x & 31;
  const int n = lane & 15, h = lane >> 4;
  const bf16_t* p = baseT + (size_t)n * ld + 8 * h;
  v8bf lo = *(const v8bf*)p;
  v8bf hi = *(const v8bf*)(p + 16);
  return frag_combine(lo, hi);
}

__device__ static inline v8f wmma_bf16(v16bf a, v16bf b, v8f c) {
  // (neg_a, A, neg_b, B, c_mod, C, reuse_a, reuse_b)
  return __builtin_amdgcn_wmma_f32_16x16x32_bf16(false, a, false, b,
                                                 (short)0, c, false, false);
}

// ---------------------------------------------------------------------------
// K1a: x[b][s][e] fp32 -> xb[b][s][e] bf16  and  xt[b][e][s] bf16 (transpose)
// ---------------------------------------------------------------------------
__global__ __launch_bounds__(256) void convert_x_kernel(const float* __restrict__ x,
                                                        bf16_t* __restrict__ xb,
                                                        bf16_t* __restrict__ xt) {
  __shared__ float tile[32][33];
  const int b  = blockIdx.z;
  const int s0 = blockIdx.x * 32;
  const int e0 = blockIdx.y * 32;
  const int tx = threadIdx.x & 31;
  const int ty = threadIdx.x >> 5;     // 0..7
#pragma unroll
  for (int i = 0; i < 4; ++i) {
    const int row = ty + 8 * i;        // s offset
    const float v = x[((size_t)b * SEQ + s0 + row) * EMB + e0 + tx];
    tile[row][tx] = v;
    xb[((size_t)b * SEQ + s0 + row) * EMB + e0 + tx] = (bf16_t)v;
  }
  __syncthreads();
#pragma unroll
  for (int i = 0; i < 4; ++i) {
    const int row = ty + 8 * i;        // e offset
    xt[((size_t)b * EMB + e0 + row) * SEQ + s0 + tx] = (bf16_t)tile[tx][row];
  }
}

// ---------------------------------------------------------------------------
// K1b: W[k][n] fp32 -> Wt[n][k] bf16 (blockIdx.z selects Wq / Wk)
// ---------------------------------------------------------------------------
__global__ __launch_bounds__(256) void convert_w_kernel(const float* __restrict__ wq,
                                                        const float* __restrict__ wk,
                                                        bf16_t* __restrict__ wqt,
                                                        bf16_t* __restrict__ wkt) {
  __shared__ float tile[32][33];
  const float* w  = blockIdx.z ? wk  : wq;
  bf16_t*      wt = blockIdx.z ? wkt : wqt;
  const int k0 = blockIdx.x * 32;
  const int n0 = blockIdx.y * 32;
  const int tx = threadIdx.x & 31;
  const int ty = threadIdx.x >> 5;
#pragma unroll
  for (int i = 0; i < 4; ++i) {
    const int row = ty + 8 * i;        // k offset
    tile[row][tx] = w[(size_t)(k0 + row) * EMB + n0 + tx];
  }
  __syncthreads();
#pragma unroll
  for (int i = 0; i < 4; ++i) {
    const int row = ty + 8 * i;        // n offset
    wt[(size_t)(n0 + row) * EMB + k0 + tx] = (bf16_t)tile[tx][row];
  }
}

// ---------------------------------------------------------------------------
// K2: projections q = xb * Wq, k = xb * Wk  (bf16 WMMA, fp32 accum, bf16 out)
// grid = (N/64, M/128, 2); 8 waves, each owns a 32x32 tile (2x2 WMMA frags).
// ---------------------------------------------------------------------------
__global__ __launch_bounds__(256) void proj_kernel(const bf16_t* __restrict__ xb,
                                                   const bf16_t* __restrict__ wqt,
                                                   const bf16_t* __restrict__ wkt,
                                                   bf16_t* __restrict__ qb,
                                                   bf16_t* __restrict__ kb) {
  const bf16_t* wt = blockIdx.z ? wkt : wqt;
  bf16_t*       op = blockIdx.z ? kb  : qb;
  const int wave = threadIdx.x >> 5;
  const int lane = threadIdx.x & 31;
  const int lrow = lane & 15, half = lane >> 4;
  const int mi = wave & 3, ni = wave >> 2;
  const int m0 = blockIdx.y * 128 + mi * 32;   // M = B*S = 16384
  const int n0 = blockIdx.x * 64 + ni * 32;    // N = 1024

  v8f c[2][2];
#pragma unroll
  for (int i = 0; i < 2; ++i)
#pragma unroll
    for (int j = 0; j < 2; ++j) c[i][j] = (v8f)(0.0f);

  for (int kk = 0; kk < EMB; kk += 32) {
    v16bf a0 = load_fragA(xb + (size_t)m0 * EMB + kk, EMB);
    v16bf a1 = load_fragA(xb + (size_t)(m0 + 16) * EMB + kk, EMB);
    v16bf b0 = load_fragB(wt + (size_t)n0 * EMB + kk, EMB);
    v16bf b1 = load_fragB(wt + (size_t)(n0 + 16) * EMB + kk, EMB);
    c[0][0] = wmma_bf16(a0, b0, c[0][0]);
    c[0][1] = wmma_bf16(a0, b1, c[0][1]);
    c[1][0] = wmma_bf16(a1, b0, c[1][0]);
    c[1][1] = wmma_bf16(a1, b1, c[1][1]);
  }
#pragma unroll
  for (int mt = 0; mt < 2; ++mt)
#pragma unroll
    for (int nt = 0; nt < 2; ++nt)
#pragma unroll
      for (int r = 0; r < 8; ++r)
        op[(size_t)(m0 + 16 * mt + r + 8 * half) * EMB + (n0 + 16 * nt + lrow)] =
            (bf16_t)c[mt][nt][r];
}

// ---------------------------------------------------------------------------
// K3: flash attention.  One workgroup = (batch, 16-query block).
// 8 waves; wave w owns d-slice [128w, 128w+128) of the fp32 O accumulator
// (8 C-frags in registers).  64-key blocks, online softmax.
// ---------------------------------------------------------------------------
__global__ __launch_bounds__(256) void attn_kernel(const bf16_t* __restrict__ qb,
                                                   const bf16_t* __restrict__ kb,
                                                   const bf16_t* __restrict__ xt,
                                                   float* __restrict__ out) {
  __shared__ __align__(16) float  spart[NWAVE][BQ][SLD]; // per-wave partial S
  __shared__ __align__(16) float  ssum[BQ][SLD];         // reduced, scaled S
  __shared__ __align__(16) bf16_t pbuf[BQ][BK];          // P (bf16) for PV

  const int b    = blockIdx.y;
  const int q0   = blockIdx.x * BQ;
  const int wave = threadIdx.x >> 5;
  const int lane = threadIdx.x & 31;
  const int lrow = lane & 15, half = lane >> 4;
  const float scale = 0.03125f;                // 1/sqrt(1024)

  const bf16_t* qbase  = qb + ((size_t)b * SEQ + q0) * EMB;
  const bf16_t* kbase  = kb + (size_t)b * SEQ * EMB;
  const bf16_t* vtbase = xt + (size_t)b * EMB * SEQ;

  // Q A-frags for this wave's embed slice [128w, 128w+128) — loop invariant.
  v16bf qa[4];
#pragma unroll
  for (int kc = 0; kc < 4; ++kc)
    qa[kc] = load_fragA(qbase + DSL * wave + 32 * kc, EMB);

  v8f o[8];
#pragma unroll
  for (int nt = 0; nt < 8; ++nt) o[nt] = (v8f)(0.0f);
  float m_run = -3.0e38f, l_run = 0.0f;

  for (int t = 0; t < SEQ / BK; ++t) {
    const int k0 = t * BK;

    // hint next key block toward the caches
    if (t + 1 < SEQ / BK)
      __builtin_prefetch(kbase + (size_t)(k0 + BK + lrow) * EMB + DSL * wave, 0, 1);

    // ---- 1) partial S over this wave's embed slice -----------------------
    v8f sc[4];
#pragma unroll
    for (int nt = 0; nt < 4; ++nt) sc[nt] = (v8f)(0.0f);
#pragma unroll
    for (int kc = 0; kc < 4; ++kc) {
#pragma unroll
      for (int nt = 0; nt < 4; ++nt) {
        v16bf bf = load_fragB(kbase + (size_t)(k0 + 16 * nt) * EMB +
                              (DSL * wave + 32 * kc), EMB);
        sc[nt] = wmma_bf16(qa[kc], bf, sc[nt]);
      }
    }
#pragma unroll
    for (int nt = 0; nt < 4; ++nt)
#pragma unroll
      for (int r = 0; r < 8; ++r)
        spart[wave][r + 8 * half][16 * nt + lrow] = sc[nt][r];
    __syncthreads();

    // ---- 2) reduce 8 partials; wave w sums columns [8w, 8w+8) ------------
    {
      const int col = 8 * wave + 4 * half;
      float ax = 0.f, ay = 0.f, az = 0.f, aw = 0.f;
#pragma unroll
      for (int p = 0; p < NWAVE; ++p) {
        const float4 v = *(const float4*)&spart[p][lrow][col];
        ax += v.x; ay += v.y; az += v.z; aw += v.w;
      }
      float4 r;
      r.x = ax * scale; r.y = ay * scale; r.z = az * scale; r.w = aw * scale;
      *(float4*)&ssum[lrow][col] = r;
    }
    __syncthreads();

    // ---- 3) online softmax (each wave redundantly; lane -> row lane%16) --
    float pv[32];
    float mx = -3.0e38f;
#pragma unroll
    for (int j = 0; j < 32; j += 4) {
      const float4 v = *(const float4*)&ssum[lrow][32 * half + j];
      pv[j + 0] = v.x; pv[j + 1] = v.y; pv[j + 2] = v.z; pv[j + 3] = v.w;
      mx = fmaxf(mx, fmaxf(fmaxf(v.x, v.y), fmaxf(v.z, v.w)));
    }
    mx = fmaxf(mx, __shfl_xor(mx, 16, 32));      // combine the two halves
    const float m_new = fmaxf(m_run, mx);
    float rs = 0.0f;
#pragma unroll
    for (int j = 0; j < 32; ++j) { pv[j] = __expf(pv[j] - m_new); rs += pv[j]; }
    rs += __shfl_xor(rs, 16, 32);
    const float alpha = __expf(m_run - m_new);
    l_run = l_run * alpha + rs;
    m_run = m_new;
    if (wave == 0) {
#pragma unroll
      for (int j = 0; j < 32; ++j) pbuf[lrow][32 * half + j] = (bf16_t)pv[j];
    }
    __syncthreads();

    // ---- 4) rescale O by alpha (per output row r + 8*half) ---------------
    float av[8];
#pragma unroll
    for (int r = 0; r < 8; ++r) av[r] = __shfl(alpha, r + 8 * half, 32);
#pragma unroll
    for (int nt = 0; nt < 8; ++nt)
#pragma unroll
      for (int r = 0; r < 8; ++r) o[nt][r] *= av[r];

    // ---- 5) O += P @ V  (V^T rows come from xt, contiguous along keys) ---
#pragma unroll
    for (int kc = 0; kc < 2; ++kc) {
      v16bf pa = load_fragA(&pbuf[0][0] + 32 * kc, BK);
#pragma unroll
      for (int nt = 0; nt < 8; ++nt) {
        v16bf vb = load_fragB(vtbase + (size_t)(DSL * wave + 16 * nt) * SEQ +
                              (k0 + 32 * kc), SEQ);
        o[nt] = wmma_bf16(pa, vb, o[nt]);
      }
    }
    __syncthreads();   // protect spart/ssum/pbuf before next iteration
  }

  // ---- epilogue: normalize by l and store fp32 output ---------------------
  const float linv = 1.0f / l_run;               // row = lane%16
  float lv[8];
#pragma unroll
  for (int r = 0; r < 8; ++r) lv[r] = __shfl(linv, r + 8 * half, 32);
  float* obase = out + ((size_t)b * SEQ + q0) * EMB;
#pragma unroll
  for (int nt = 0; nt < 8; ++nt)
#pragma unroll
    for (int r = 0; r < 8; ++r)
      obase[(size_t)(r + 8 * half) * EMB + DSL * wave + 16 * nt + lrow] =
          o[nt][r] * lv[r];
}

// ---------------------------------------------------------------------------
extern "C" void kernel_launch(void* const* d_in, const int* in_sizes, int n_in,
                              void* d_out, int out_size, void* d_ws, size_t ws_size,
                              hipStream_t stream) {
  (void)in_sizes; (void)n_in; (void)out_size; (void)ws_size;
  const float* x  = (const float*)d_in[0];
  const float* Wq = (const float*)d_in[1];
  const float* Wk = (const float*)d_in[2];
  float* out = (float*)d_out;

  // bf16 staging in workspace (132 MB total)
  const size_t NX = (size_t)BATCH * SEQ * EMB;     // 16M elements
  bf16_t* xb  = (bf16_t*)d_ws;
  bf16_t* xt  = xb  + NX;
  bf16_t* wqt = xt  + NX;
  bf16_t* wkt = wqt + (size_t)EMB * EMB;
  bf16_t* qb  = wkt + (size_t)EMB * EMB;
  bf16_t* kb  = qb  + NX;

  convert_x_kernel<<<dim3(SEQ / 32, EMB / 32, BATCH), 256, 0, stream>>>(x, xb, xt);
  convert_w_kernel<<<dim3(EMB / 32, EMB / 32, 2), 256, 0, stream>>>(Wq, Wk, wqt, wkt);
  proj_kernel<<<dim3(EMB / 64, (BATCH * SEQ) / 128, 2), 256, 0, stream>>>(xb, wqt, wkt, qb, kb);
  attn_kernel<<<dim3(SEQ / BQ, BATCH), 256, 0, stream>>>(qb, kb, xt, out);
}
